// SparseConv3dTestTorch_45311904972969
// MI455X (gfx1250) — matrix operand order repaired and checked by reading, not compiled, not run on hardware
//
#include <hip/hip_runtime.h>

#define CIN 32
#define COUT 64
#define KS 3
#define STR 2
#define PAD 1
#define DOUT 48
#define NOFF 27                       // 3*3*3 kernel offsets
#define PWELEMS (NOFF * 4 * 32 * 16)  // packed bf16 W: offsets * colblocks * lanes * 16 halves

typedef __bf16          v16bf __attribute__((ext_vector_type(16)));
typedef unsigned short  v16us __attribute__((ext_vector_type(16)));
typedef float           v8f   __attribute__((ext_vector_type(8)));

__device__ __forceinline__ unsigned short f2bf(float f) {
    unsigned int u = __builtin_bit_cast(unsigned int, f);
    unsigned int r = u + 0x7FFFu + ((u >> 16) & 1u);   // round-to-nearest-even
    return (unsigned short)(r >> 16);
}

// ---------------------------------------------------------------------------
// Zero-fill output (scatter-add target), vectorized 16B stores.
// ---------------------------------------------------------------------------
__global__ void zero_out4(float4* __restrict__ out, int n4) {
    int i = blockIdx.x * blockDim.x + threadIdx.x;
    if (i < n4) out[i] = make_float4(0.f, 0.f, 0.f, 0.f);
}

// ---------------------------------------------------------------------------
// Pre-pack W [27][32][64] fp32 -> bf16 in exact WMMA B-fragment register
// layout: pw[((ko*4+cb)*32 + lane)*16 + j] = bf16(Wk[(hi*16+j)*64 + cb*16+col])
// where hi = lane>>4, col = lane&15. One 32B load per B fragment in hot loop.
// ---------------------------------------------------------------------------
__global__ void pack_w_bf16(const float* __restrict__ W, unsigned short* __restrict__ pw) {
    int t = blockIdx.x * blockDim.x + threadIdx.x;
    if (t >= PWELEMS) return;
    int j    = t & 15;
    int lane = (t >> 4) & 31;
    int cb   = (t >> 9) & 3;
    int ko   = t >> 11;
    int hi = lane >> 4, col = lane & 15;
    float v = W[(ko * CIN + (hi * 16 + j)) * COUT + cb * 16 + col];
    pw[t] = f2bf(v);
}

// ---------------------------------------------------------------------------
// A-fragment loader: 16x32 bf16 tile for voxels [base, base+16).
// Lane layout: lanes 0-15 hold K=0..15; lanes 16-31 hold K=16..31 (element j = K 16*hi+j).
// Also returns this lane's mirrored-row coordinates.
// ---------------------------------------------------------------------------
__device__ __forceinline__ v16bf load_A(const float* __restrict__ feats,
                                        const int* __restrict__ coors,
                                        int base, int row, int hi, int nvox,
                                        int& cb0, int& cz, int& cy, int& cx, bool& vok)
{
    const int vox = base + row;
    vok = vox < nvox;
    const int vc = vok ? vox : (nvox - 1);
    cb0 = coors[vc * 4 + 0];
    cz  = coors[vc * 4 + 1];
    cy  = coors[vc * 4 + 2];
    cx  = coors[vc * 4 + 3];
    v16us aus;
    const float* fr = feats + (size_t)vc * CIN;
    const int c0 = hi * 16;
    const int c1 = hi * 16 + 8;
#pragma unroll
    for (int j = 0; j < 8; ++j) {
        float lo  = vok ? fr[c0 + j] : 0.f;
        float hi2 = vok ? fr[c1 + j] : 0.f;
        aus[j]     = f2bf(lo);
        aus[8 + j] = f2bf(hi2);
    }
    return __builtin_bit_cast(v16bf, aus);
}

__device__ __forceinline__ int out_index(int cb0, int cz, int cy, int cx,
                                         int kz, int ky, int kx, bool vok)
{
    const int zn = cz + PAD - kz;
    const int yn = cy + PAD - ky;
    const int xn = cx + PAD - kx;
    bool valid = vok && !(zn & 1) && !(yn & 1) && !(xn & 1);
    const int zo = zn >> 1, yo = yn >> 1, xo = xn >> 1;
    valid = valid && ((unsigned)zo < DOUT) && ((unsigned)yo < DOUT) && ((unsigned)xo < DOUT);
    return valid ? (((cb0 * DOUT + zo) * DOUT + yo) * DOUT + xo) : -1;
}

// ---------------------------------------------------------------------------
// Main kernel: one wave32 per 32-voxel super-tile (two 16-row A fragments).
// Per offset: 4 shared B fragments feed 8 v_wmma_f32_16x16x32_bf16, then
// masked atomic scatter-add. PACKED selects single-load B vs on-the-fly cvt.
// ---------------------------------------------------------------------------
template <bool PACKED>
__global__ __launch_bounds__(256) void spconv_wmma(
    const float* __restrict__ feats,
    const int*   __restrict__ coors,
    const float* __restrict__ W,
    const unsigned short* __restrict__ packedW,
    float* __restrict__ out,
    int nvox)
{
    const int lane  = threadIdx.x & 31;
    const int wave  = blockIdx.x * (blockDim.x >> 5) + (threadIdx.x >> 5);
    const int ntile = (nvox + 31) >> 5;
    if (wave >= ntile) return;                     // wave-uniform exit

    const int row = lane & 15;
    const int hi  = lane >> 4;
    const int base = wave * 32;

    int b0, z0, y0, x0, b1, z1, y1, x1;
    bool vok0, vok1;
    const v16bf A0 = load_A(feats, coors, base,      row, hi, nvox, b0, z0, y0, x0, vok0);
    const v16bf A1 = load_A(feats, coors, base + 16, row, hi, nvox, b1, z1, y1, x1, vok1);

    const v16us* pwv = (const v16us*)packedW;

    for (int ko = 0; ko < NOFF; ++ko) {
        const int kz = ko / 9, ky = (ko / 3) % 3, kx = ko % 3;
        const int oidx0 = out_index(b0, z0, y0, x0, kz, ky, kx, vok0);
        const int oidx1 = out_index(b1, z1, y1, x1, kz, ky, kx, vok1);

        // Uniformly skip offsets where every row of both tiles is invalid.
        const unsigned long long bal = __ballot((oidx0 >= 0) || (oidx1 >= 0));
        if (bal == 0ull) continue;

        // ---- 4 column blocks; each B fragment feeds two WMMAs (M=32) ----
        v8f acc0[4], acc1[4];
#pragma unroll
        for (int cb = 0; cb < 4; ++cb) {
            v16us bus;
            if (PACKED) {
                bus = pwv[(ko * 4 + cb) * 32 + lane];          // one 32B load
            } else {
                const float* wb = W + (size_t)ko * CIN * COUT + (hi * 16) * COUT + cb * 16 + row;
#pragma unroll
                for (int j = 0; j < 16; ++j) bus[j] = f2bf(wb[j * COUT]);
            }
            const v16bf B = __builtin_bit_cast(v16bf, bus);
            v8f c0 = {}, c1 = {};
            acc0[cb] = __builtin_amdgcn_wmma_f32_16x16x32_bf16(
                false, A0, false, B, (short)0, c0, false, false);
            acc1[cb] = __builtin_amdgcn_wmma_f32_16x16x32_bf16(
                false, A1, false, B, (short)0, c1, false, false);
        }

        // ---- Masked scatter-add. D layout: VGPR v -> row v + 8*hi, col = lane&15 ----
#pragma unroll
        for (int v = 0; v < 8; ++v) {
            const int r = v + hi * 8;
            const int r0 = __shfl(oidx0, r, 32);
            if (r0 >= 0) {
                float* orow = out + (size_t)r0 * COUT;
#pragma unroll
                for (int cb = 0; cb < 4; ++cb)
                    atomicAdd(orow + cb * 16 + row, acc0[cb][v]);
            }
            const int r1 = __shfl(oidx1, r, 32);
            if (r1 >= 0) {
                float* orow = out + (size_t)r1 * COUT;
#pragma unroll
                for (int cb = 0; cb < 4; ++cb)
                    atomicAdd(orow + cb * 16 + row, acc1[cb][v]);
            }
        }
    }
}

// ---------------------------------------------------------------------------
// Host-side launcher. Inputs: [0]=features f32[N,32], [1]=coors i32[N,4],
// [2]=batch_size (scalar, unused), [3]=W f32[3,3,3,32,64]. Output f32.
// ---------------------------------------------------------------------------
extern "C" void kernel_launch(void* const* d_in, const int* in_sizes, int n_in,
                              void* d_out, int out_size, void* d_ws, size_t ws_size,
                              hipStream_t stream) {
    const float* feats = (const float*)d_in[0];
    const int*   coors = (const int*)d_in[1];
    const float* W     = (const float*)d_in[3];
    float* out = (float*)d_out;

    const int nvox = in_sizes[0] / CIN;

    // 1) zero the scatter-add target
    const int n4 = out_size / 4;
    zero_out4<<<(n4 + 255) / 256, 256, 0, stream>>>((float4*)out, n4);

    // 2) pre-pack W to bf16 in WMMA B-fragment layout (if workspace fits)
    const bool packed = ws_size >= (size_t)PWELEMS * sizeof(unsigned short);
    unsigned short* pw = nullptr;
    if (packed) {
        pw = (unsigned short*)d_ws;
        pack_w_bf16<<<(PWELEMS + 255) / 256, 256, 0, stream>>>(W, pw);
    }

    // 3) fused gather-GEMM-scatter: one wave per 32-voxel super-tile
    const int ntile  = (nvox + 31) / 32;
    const int blocks = (ntile + 7) / 8;
    if (packed)
        spconv_wmma<true><<<blocks, 256, 0, stream>>>(feats, coors, W, pw, out, nvox);
    else
        spconv_wmma<false><<<blocks, 256, 0, stream>>>(feats, coors, W, pw, out, nvox);
}
